// DBN_33483565039734
// MI455X (gfx1250) — matrix operand order repaired and checked by reading, not compiled, not run on hardware
//
#include <hip/hip_runtime.h>
#include <math.h>

#define B_ 4
#define L_ 256
#define V_ 32000
#define K_ 16
#define E_ 128
#define H_ 256
#define NSTEP 3
#define NSAMPLE 2
#define LR 0.01f
#define BETASQ 1.0f
#define BETASQ2 10.0f
#define SIGMA 0.316227766016838f
#define C1 (-1.15129254649702f)   /* 0.5*(LOG_A-LOG_B) = -0.5*ln(10) */

#define ROWS (B_*L_)              /* 1024 */
#define LE_  (L_*E_)              /* 32768 */
#define RT_N (ROWS/16)            /* 64 row tiles */
#define NCHF 40                   /* fwd vocab chunks */
#define VTPC (V_/16/NCHF)         /* 50 v-tiles per fwd chunk */
#define NCHB 10                   /* bwd vocab chunks */
#define GPCB (V_/32/NCHB)         /* 100 32-v groups per bwd chunk */

typedef __attribute__((ext_vector_type(16))) __bf16 v16bf;
typedef __attribute__((ext_vector_type(8)))  float  v8f;
typedef __attribute__((ext_vector_type(8)))  int    v8i;

static __device__ __forceinline__ unsigned f32_to_bf16(float f) {
  unsigned u = __float_as_uint(f);
  return (u + 0x7FFFu + ((u >> 16) & 1u)) >> 16;   // RNE
}
static __device__ __forceinline__ unsigned short f32_to_bf16_fast(float f) {
  return (unsigned short)((__float_as_uint(f) + 0x8000u) >> 16);
}

// async copy of one contiguous 512B wave-slice; IOFFSET applies to both sides
#define ASYNC_B128(lofs, gaddr, OFS)                                          \
  asm volatile("global_load_async_to_lds_b128 %0, %1, off offset:" OFS        \
               :: "v"(lofs), "v"(gaddr) : "memory")

// copy 4KB (16 vocab rows x 128 e, bf16, contiguous) global -> LDS, async
static __device__ __forceinline__ void async_tile_4k(unsigned lofs,
                                                     unsigned long long g) {
  ASYNC_B128(lofs, g, "0");
  ASYNC_B128(lofs, g, "512");
  ASYNC_B128(lofs, g, "1024");
  ASYNC_B128(lofs, g, "1536");
  ASYNC_B128(lofs, g, "2048");
  ASYNC_B128(lofs, g, "2560");
  ASYNC_B128(lofs, g, "3072");
  ASYNC_B128(lofs, g, "3584");
}

// load a 32-byte bf16 A/B fragment as two aligned 16B chunks
static __device__ __forceinline__ v16bf frag_2x16(const unsigned short* lo,
                                                  const unsigned short* hi) {
  int4 a = *(const int4*)lo;
  int4 b = *(const int4*)hi;
  v8i pk;
  pk[0] = a.x; pk[1] = a.y; pk[2] = a.z; pk[3] = a.w;
  pk[4] = b.x; pk[5] = b.y; pk[6] = b.z; pk[7] = b.w;
  return __builtin_bit_cast(v16bf, pk);
}

// ---------------------------------------------------------------- utilities
__global__ void dbn_zero(float* __restrict__ p, int n) {
  int i = blockIdx.x * blockDim.x + threadIdx.x;
  if (i < n) p[i] = 0.0f;
}
__global__ void dbn_copy(float* __restrict__ dst, const float* __restrict__ src, int n) {
  int i = blockIdx.x * blockDim.x + threadIdx.x;
  if (i < n) dst[i] = src[i];
}
__global__ void dbn_write(const float* __restrict__ lpb, float* __restrict__ out) {
  int i = threadIdx.x;
  if (i < B_) out[i] = lpb[i];
}
__global__ void dbn_update(float* __restrict__ nh, const float* __restrict__ gacc) {
  int i = blockIdx.x * blockDim.x + threadIdx.x;
  if (i < B_ * K_ * H_) nh[i] += (LR / (float)NSAMPLE) * gacc[i];
}

// ------------------------------------- one-time bf16 packing of emb_vocab
__global__ void dbn_pack_emb(const float* __restrict__ emb,
                             unsigned short* __restrict__ embh) {
  int i = blockIdx.x * blockDim.x + threadIdx.x;
  if (i < V_ * E_) embh[i] = (unsigned short)f32_to_bf16(emb[i]);
}
__global__ void dbn_pack_embT(const float* __restrict__ emb,
                              unsigned short* __restrict__ embT) {
  int i = blockIdx.x * blockDim.x + threadIdx.x;
  if (i >= V_ * E_) return;
  int e = i / V_, v = i - e * V_;                 // contiguous writes over v
  embT[i] = (unsigned short)f32_to_bf16(emb[(size_t)v * E_ + e]);
}

// ------------------------------------------------- noise + perturbed state
__global__ void dbn_noise(const float* __restrict__ nh, float* __restrict__ noise,
                          float* __restrict__ nhn, int step, int smp) {
  int i = blockIdx.x * blockDim.x + threadIdx.x;
  if (i >= B_ * K_ * H_) return;
  unsigned long long z = (((unsigned long long)(step * NSAMPLE + smp + 1)) << 34)
                         ^ (unsigned long long)i;
  z += 0x9E3779B97F4A7C15ull;
  z = (z ^ (z >> 30)) * 0xBF58476D1CE4E5B9ull;
  z = (z ^ (z >> 27)) * 0x94D049BB133111EBull;
  z ^= z >> 31;
  unsigned u1 = (unsigned)z, u2 = (unsigned)(z >> 32);
  float f1 = ((float)u1 + 1.0f) * 2.3283064365e-10f;
  float f2 = (float)u2 * 2.3283064365e-10f;
  float nrm = sqrtf(-2.0f * logf(f1)) * __cosf(6.28318530718f * f2);
  float nz = SIGMA * nrm;
  noise[i] = nz;
  nhn[i] = nh[i] + nz;
}

// ---------------------------------------------- wke = sigmoid(nhn[:,1:]@w_k+b_k)
__global__ __launch_bounds__(256) void dbn_wke(const float* __restrict__ nhn,
                                               const float* __restrict__ w_k,
                                               const float* __restrict__ b_k,
                                               float* __restrict__ wke) {
  __shared__ float row[H_];
  int b = blockIdx.x, kk = blockIdx.y, j = threadIdx.x;
  row[j] = nhn[((size_t)b * K_ + kk + 1) * H_ + j];
  __syncthreads();
  const float* w = w_k + (size_t)kk * H_ * H_;
  float z = b_k[kk * H_ + j];
  for (int h = 0; h < H_; ++h) z += row[h] * w[h * H_ + j];
  wke[((size_t)b * (K_ - 1) + kk) * H_ + j] = 1.0f / (1.0f + __expf(-z));
}

// -------------- residual r, direct grad, sigmoid backprop seed t, lp_internal
__global__ __launch_bounds__(256) void dbn_internal(const float* __restrict__ nhn,
                                                    const float* __restrict__ noise,
                                                    const float* __restrict__ wke,
                                                    float* __restrict__ t,
                                                    float* __restrict__ gacc,
                                                    float* __restrict__ lpb) {
  __shared__ float red[256];
  int b = blockIdx.x, k = blockIdx.y, h = threadIdx.x;
  size_t idx = ((size_t)b * K_ + k) * H_ + h;
  float w = 0.0f, dsig = 0.0f;
  if (k < K_ - 1) {
    w = wke[((size_t)b * (K_ - 1) + k) * H_ + h];
    dsig = w * (1.0f - w);
  }
  float r = nhn[idx] - w;
  gacc[idx] += -BETASQ * r;
  if (k < K_ - 1) t[((size_t)b * (K_ - 1) + k) * H_ + h] = BETASQ * r * dsig;
  float nz = noise[idx];
  red[h] = C1 - 0.5f * BETASQ * r * r + 0.5f * BETASQ2 * nz * nz;
  __syncthreads();
  for (int s = 128; s > 0; s >>= 1) { if (h < s) red[h] += red[h + s]; __syncthreads(); }
  if (h == 0) atomicAdd(&lpb[b], red[0] * (1.0f / (float)NSAMPLE));
}

// ------------------------- grad through sigmoid: row k gets t[k-1] @ w_k[k-1]^T
__global__ __launch_bounds__(256) void dbn_gradsig(const float* __restrict__ t,
                                                   const float* __restrict__ w_k,
                                                   float* __restrict__ gacc) {
  __shared__ float trow[H_];
  int b = blockIdx.x, kk = blockIdx.y, h = threadIdx.x;
  trow[h] = t[((size_t)b * (K_ - 1) + kk) * H_ + h];
  __syncthreads();
  const float* wrow = w_k + ((size_t)kk * H_ + h) * H_;
  float acc = 0.0f;
  for (int j = 0; j < H_; ++j) acc += trow[j] * wrow[j];
  gacc[((size_t)b * K_ + kk + 1) * H_ + h] += acc;
}

// ----------------- node_e = nhn[:,0] @ w_hle + b_hle   (f32 + bf16 copies)
__global__ __launch_bounds__(256) void dbn_node_e(const float* __restrict__ nhn,
                                                  const float* __restrict__ w_hle,
                                                  const float* __restrict__ b_hle,
                                                  float* __restrict__ node_e,
                                                  unsigned short* __restrict__ nodeh) {
  __shared__ float n0[H_];
  int b = blockIdx.x, tid = threadIdx.x;
  n0[tid] = nhn[(size_t)b * K_ * H_ + tid];
  __syncthreads();
  int le = blockIdx.y * 256 + tid;
  float acc = b_hle[le];
  for (int h = 0; h < H_; ++h) acc += n0[h] * w_hle[(size_t)h * LE_ + le];
  node_e[(size_t)b * LE_ + le] = acc;
  nodeh[(size_t)b * LE_ + le] = (unsigned short)f32_to_bf16(acc);
}

// --------------- forward WMMA GEMM over vocab with online softmax statistics
// emb tiles streamed global -> LDS with async b128 copies, double buffered
__global__ __launch_bounds__(32) void dbn_fwd_gemm(const unsigned short* __restrict__ nodeh,
                                                   const unsigned short* __restrict__ embh,
                                                   float* __restrict__ parts) {
  __shared__ __align__(32) unsigned short tiles[2][16 * 128];
  const int rt = blockIdx.x, ch = blockIdx.y;
  const int lane = threadIdx.x, grp = lane >> 4, nidx = lane & 15;
  const int R0 = rt * 16;

  // resident A fragments: node_e rows (M = b,l row; K = e)
  v16bf afrag[4];
#pragma unroll
  for (int ks = 0; ks < 4; ++ks) {
    const unsigned short* base = nodeh + (size_t)(R0 + nidx) * E_ + ks * 32 + grp * 16;
    afrag[ks] = frag_2x16(base, base + 8);
  }

  float m[8], s[8];
#pragma unroll
  for (int r = 0; r < 8; ++r) { m[r] = -3.0e38f; s[r] = 0.0f; }

  const int v_begin = ch * VTPC * 16;
  const unsigned lb0 = (unsigned)(size_t)&tiles[0][0];
  const unsigned lb1 = (unsigned)(size_t)&tiles[1][0];
  {
    unsigned lofs = lb0 + (unsigned)lane * 16u;
    unsigned long long g = (unsigned long long)(size_t)(embh + (size_t)v_begin * E_)
                           + (unsigned long long)lane * 16ull;
    async_tile_4k(lofs, g);
  }
  int buf = 0;
  for (int t = 0; t < VTPC; ++t) {
    if (t + 1 < VTPC) {
      unsigned lofs = (buf ? lb0 : lb1) + (unsigned)lane * 16u;
      unsigned long long g =
          (unsigned long long)(size_t)(embh + (size_t)(v_begin + (t + 1) * 16) * E_)
          + (unsigned long long)lane * 16ull;
      async_tile_4k(lofs, g);
      asm volatile("s_wait_asynccnt 0x8" ::: "memory");
    } else {
      asm volatile("s_wait_asynccnt 0x0" ::: "memory");
    }
    const unsigned short* tp = &tiles[buf][0];
    v8f c;
#pragma unroll
    for (int r = 0; r < 8; ++r) c[r] = 0.0f;
#pragma unroll
    for (int ks = 0; ks < 4; ++ks) {
      v16bf bfrag = *(const v16bf*)(tp + nidx * 128 + ks * 32 + grp * 16);
      c = __builtin_amdgcn_wmma_f32_16x16x32_bf16(false, afrag[ks], false, bfrag,
                                                  (short)0, c, false, false);
    }
    // per-lane online softmax (cross-lane merge deferred to chunk end)
#pragma unroll
    for (int r = 0; r < 8; ++r) {
      float cv = c[r];
      float nm = fmaxf(m[r], cv);
      s[r] = s[r] * __expf(m[r] - nm) + __expf(cv - nm);
      m[r] = nm;
    }
    buf ^= 1;
  }
  // merge the 16 per-lane partials of each row (log-sum-exp combine)
#pragma unroll
  for (int r = 0; r < 8; ++r) {
    float mm = m[r], ss = s[r];
#pragma unroll
    for (int d = 1; d <= 8; d <<= 1) {
      float om = __shfl_xor(mm, d, 32);
      float os = __shfl_xor(ss, d, 32);
      float nm = fmaxf(mm, om);
      ss = ss * __expf(mm - nm) + os * __expf(om - nm);
      mm = nm;
    }
    m[r] = mm; s[r] = ss;
  }
  if (nidx == 0) {
#pragma unroll
    for (int r = 0; r < 8; ++r) {
      size_t o = ((size_t)(rt * NCHF + ch) * 16 + r + grp * 8) * 2;
      parts[o] = m[r];
      parts[o + 1] = s[r];
    }
  }
}

// --------------------------- combine per-chunk stats -> final (max, sumexp, lse)
__global__ __launch_bounds__(256) void dbn_combine(const float* __restrict__ parts,
                                                   float* __restrict__ fin,
                                                   float* __restrict__ lse) {
  int row = blockIdx.x * 256 + threadIdx.x;
  if (row >= ROWS) return;
  int rt = row >> 4, rr = row & 15;
  float gm = -3.0e38f;
  for (int c = 0; c < NCHF; ++c)
    gm = fmaxf(gm, parts[((size_t)(rt * NCHF + c) * 16 + rr) * 2]);
  float s = 0.0f;
  for (int c = 0; c < NCHF; ++c) {
    size_t o = ((size_t)(rt * NCHF + c) * 16 + rr) * 2;
    s += parts[o + 1] * __expf(parts[o] - gm);
  }
  fin[row * 2] = gm;
  fin[row * 2 + 1] = s;
  lse[row] = gm + logf(s);
}

// --------- logit at tok + accumulate lp_external = sum_l (logit_tok - lse)
__global__ __launch_bounds__(128) void dbn_lptok(const float* __restrict__ node_e,
                                                 const float* __restrict__ emb,
                                                 const int* __restrict__ tok,
                                                 const float* __restrict__ lse,
                                                 float* __restrict__ lpb) {
  __shared__ float red[128];
  int row = blockIdx.x, e = threadIdx.x;
  int tv = tok[row];
  red[e] = node_e[(size_t)row * E_ + e] * emb[(size_t)tv * E_ + e];
  __syncthreads();
  for (int s = 64; s > 0; s >>= 1) { if (e < s) red[e] += red[e + s]; __syncthreads(); }
  if (e == 0) atomicAdd(&lpb[row >> 8], (red[0] - lse[row]) * (1.0f / (float)NSAMPLE));
}

// ---------------- backward: recompute logits, p = softmax, pe += p @ emb (WMMA)
__global__ __launch_bounds__(32) void dbn_bwd_gemm(const unsigned short* __restrict__ nodeh,
                                                   const unsigned short* __restrict__ embh,
                                                   const unsigned short* __restrict__ embT,
                                                   const float* __restrict__ fin,
                                                   float* __restrict__ pe) {
  __shared__ __align__(16) unsigned short pldh[16][40];   // 16 rows x 32 v, bf16
  const int rt = blockIdx.x, ch = blockIdx.y;
  const int lane = threadIdx.x, grp = lane >> 4, nidx = lane & 15;
  const int R0 = rt * 16;

  v16bf afrag[4];
#pragma unroll
  for (int ks = 0; ks < 4; ++ks) {
    const unsigned short* base = nodeh + (size_t)(R0 + nidx) * E_ + ks * 32 + grp * 16;
    afrag[ks] = frag_2x16(base, base + 8);
  }

  float mr[8], si[8];
#pragma unroll
  for (int r = 0; r < 8; ++r) {
    int row = R0 + r + grp * 8;
    mr[r] = fin[row * 2];
    si[r] = 1.0f / fin[row * 2 + 1];
  }

  v8f acc[8];
#pragma unroll
  for (int et = 0; et < 8; ++et)
#pragma unroll
    for (int r = 0; r < 8; ++r) acc[et][r] = 0.0f;

  const int g_begin = ch * GPCB;
  for (int g = 0; g < GPCB; ++g) {
    const int vg = (g_begin + g) * 32;
    // recompute two 16-v logit tiles, convert to probabilities in LDS (bf16)
#pragma unroll
    for (int half = 0; half < 2; ++half) {
      const int v0 = vg + half * 16;
      v8f c;
#pragma unroll
      for (int r = 0; r < 8; ++r) c[r] = 0.0f;
#pragma unroll
      for (int ks = 0; ks < 4; ++ks) {
        v16bf bfrag = *(const v16bf*)(embh + (size_t)(v0 + nidx) * E_ + ks * 32 + grp * 16);
        c = __builtin_amdgcn_wmma_f32_16x16x32_bf16(false, afrag[ks], false, bfrag,
                                                    (short)0, c, false, false);
      }
#pragma unroll
      for (int r = 0; r < 8; ++r)
        pldh[r + grp * 8][half * 16 + nidx] =
            f32_to_bf16_fast(__expf(c[r] - mr[r]) * si[r]);
    }
    __syncthreads();
    // re-fragment P (16 rows x 32 v) as WMMA A operand: contiguous bf16 chunks
    v16bf pfrag = frag_2x16(&pldh[nidx][grp * 16], &pldh[nidx][grp * 16 + 8]);
    __syncthreads();
    // pe[row, e] += sum_v P[row,v] * emb[v,e]   (K = v contiguous in embT)
#pragma unroll
    for (int et = 0; et < 8; ++et) {
      v16bf bfrag = *(const v16bf*)(embT + (size_t)(et * 16 + nidx) * V_ + vg + grp * 16);
      acc[et] = __builtin_amdgcn_wmma_f32_16x16x32_bf16(false, pfrag, false, bfrag,
                                                        (short)0, acc[et], false, false);
    }
  }
#pragma unroll
  for (int et = 0; et < 8; ++et)
#pragma unroll
    for (int r = 0; r < 8; ++r)
      atomicAdd(&pe[(size_t)(R0 + r + grp * 8) * E_ + et * 16 + nidx], acc[et][r]);
}

// ---------------------- dE = emb[tok] - pe  (d lp_ext / d node_e), in place
__global__ void dbn_dE(float* __restrict__ pe, const float* __restrict__ emb,
                       const int* __restrict__ tok) {
  int i = blockIdx.x * blockDim.x + threadIdx.x;
  if (i >= ROWS * E_) return;
  int row = i >> 7, e = i & 127;
  pe[i] = emb[(size_t)tok[row] * E_ + e] - pe[i];
}

// --------------- d lp_ext / d nhn[b,0,h] = sum_le dE[b,le] * w_hle[h,le]
__global__ __launch_bounds__(256) void dbn_gradh0(const float* __restrict__ dE,
                                                  const float* __restrict__ w_hle,
                                                  float* __restrict__ gacc) {
  __shared__ float red[256];
  int b = blockIdx.x >> 8, h = blockIdx.x & 255, tid = threadIdx.x;
  const float* wrow = w_hle + (size_t)h * LE_;
  const float* de = dE + (size_t)b * LE_;
  float acc = 0.0f;
  for (int le = tid; le < LE_; le += 256) acc += de[le] * wrow[le];
  red[tid] = acc;
  __syncthreads();
  for (int s = 128; s > 0; s >>= 1) { if (tid < s) red[tid] += red[tid + s]; __syncthreads(); }
  if (tid == 0) gacc[(size_t)b * K_ * H_ + h] += red[0];
}

// ============================================================== host driver
extern "C" void kernel_launch(void* const* d_in, const int* in_sizes, int n_in,
                              void* d_out, int out_size, void* d_ws, size_t ws_size,
                              hipStream_t stream) {
  const int*   tok   = (const int*)d_in[0];
  const float* nh0   = (const float*)d_in[1];
  const float* emb   = (const float*)d_in[2];
  const float* w_k   = (const float*)d_in[3];
  const float* b_k   = (const float*)d_in[4];
  const float* w_hle = (const float*)d_in[5];
  const float* b_hle = (const float*)d_in[6];
  float* out = (float*)d_out;

  float* w = (float*)d_ws;
  float* nh     = w; w += B_ * K_ * H_;
  float* noise  = w; w += B_ * K_ * H_;
  float* nhn    = w; w += B_ * K_ * H_;
  float* wke    = w; w += B_ * (K_ - 1) * H_;
  float* tbuf   = w; w += B_ * (K_ - 1) * H_;
  float* gacc   = w; w += B_ * K_ * H_;
  float* node_e = w; w += ROWS * E_;
  float* pe     = w; w += ROWS * E_;
  float* parts  = w; w += RT_N * NCHF * 16 * 2;
  float* fin    = w; w += ROWS * 2;
  float* lse    = w; w += ROWS;
  float* lpb    = w; w += 16;
  unsigned short* embh  = (unsigned short*)w; w += (V_ * E_) / 2;
  unsigned short* embT  = (unsigned short*)w; w += (V_ * E_) / 2;
  unsigned short* nodeh = (unsigned short*)w; w += (ROWS * E_) / 2;

  dbn_pack_emb<<<(V_ * E_) / 256, 256, 0, stream>>>(emb, embh);
  dbn_pack_embT<<<(V_ * E_) / 256, 256, 0, stream>>>(emb, embT);
  dbn_copy<<<64, 256, 0, stream>>>(nh, nh0, B_ * K_ * H_);

  for (int step = 0; step < NSTEP; ++step) {
    dbn_zero<<<64, 256, 0, stream>>>(gacc, B_ * K_ * H_);
    dbn_zero<<<1, 32, 0, stream>>>(lpb, 16);
    for (int smp = 0; smp < NSAMPLE; ++smp) {
      dbn_noise<<<64, 256, 0, stream>>>(nh, noise, nhn, step, smp);
      dbn_wke<<<dim3(B_, K_ - 1), 256, 0, stream>>>(nhn, w_k, b_k, wke);
      dbn_internal<<<dim3(B_, K_), 256, 0, stream>>>(nhn, noise, wke, tbuf, gacc, lpb);
      dbn_gradsig<<<dim3(B_, K_ - 1), 256, 0, stream>>>(tbuf, w_k, gacc);
      dbn_node_e<<<dim3(B_, LE_ / 256), 256, 0, stream>>>(nhn, w_hle, b_hle, node_e, nodeh);
      dbn_fwd_gemm<<<dim3(RT_N, NCHF), 32, 0, stream>>>(nodeh, embh, parts);
      dbn_combine<<<ROWS / 256, 256, 0, stream>>>(parts, fin, lse);
      dbn_lptok<<<ROWS, 128, 0, stream>>>(node_e, emb, tok, lse, lpb);
      dbn_zero<<<ROWS * E_ / 256, 256, 0, stream>>>(pe, ROWS * E_);
      dbn_bwd_gemm<<<dim3(RT_N, NCHB), 32, 0, stream>>>(nodeh, embh, embT, fin, pe);
      dbn_dE<<<ROWS * E_ / 256, 256, 0, stream>>>(pe, emb, tok);
      dbn_gradh0<<<B_ * H_, 256, 0, stream>>>(pe, w_hle, gacc);
    }
    dbn_update<<<64, 256, 0, stream>>>(nh, gacc);
  }
  dbn_write<<<1, B_, 0, stream>>>(lpb, out);

  (void)in_sizes; (void)n_in; (void)out_size; (void)ws_size;
}
